// SAGE_26268019983001
// MI455X (gfx1250) — matrix operand order repaired and checked by reading, not compiled, not run on hardware
//
#include <hip/hip_runtime.h>

typedef __attribute__((ext_vector_type(16))) _Float16 v16h;
typedef __attribute__((ext_vector_type(8)))  float    v8f;

#define BN_EPS 1e-5f

// ---------------------------------------------------------------------------
// Branch-free 4-element chunk load (f32 -> f16). Chunk is either fully
// loadable as one 16B float4, or partially/fully out of the K range
// (folds to zeros / guarded scalars only at the K tail).
// ---------------------------------------------------------------------------
template <int KMAX>
__device__ __forceinline__ void load_chunk(const float* __restrict__ p, int cs,
                                           _Float16* t) {
  if (cs + 3 < KMAX) {
    float4 v = *(const float4*)(p + cs);
    t[0] = (_Float16)v.x; t[1] = (_Float16)v.y;
    t[2] = (_Float16)v.z; t[3] = (_Float16)v.w;
  } else if (cs < KMAX) {
#pragma unroll
    for (int i = 0; i < 4; ++i)
      t[i] = (cs + i < KMAX) ? (_Float16)p[cs + i] : (_Float16)0.0f;
  } else {
#pragma unroll
    for (int i = 0; i < 4; ++i) t[i] = (_Float16)0.0f;
  }
}

// A-fragment (16x32 f16): lane half 0 -> K octets [k0..k0+7],[k0+16..k0+23],
// half 1 -> +8. p = row base pointer (row already clamped to valid range).
template <int KMAX>
__device__ __forceinline__ v16h load_a_f32(const float* __restrict__ p, int kb,
                                           int half) {
  _Float16 t[16];
  int k0 = kb * 32 + half * 8;
  load_chunk<KMAX>(p, k0,      t + 0);
  load_chunk<KMAX>(p, k0 + 4,  t + 4);
  load_chunk<KMAX>(p, k0 + 16, t + 8);
  load_chunk<KMAX>(p, k0 + 20, t + 12);
  v16h a;
#pragma unroll
  for (int i = 0; i < 16; ++i) a[i] = t[i];
  return a;
}

// ---------------------------------------------------------------------------
// Stage W [ncols x kmax] (row-major, ldw) into LDS as f16 in WMMA B-fragment
// order: Blds[((kb*2+half)*128 + col)*16 + j] = W[col][kb*32+half*8+(j&7)+(j>>3)*16]
// -> each lane's 16-half B fragment is one contiguous 32-byte LDS read.
// ---------------------------------------------------------------------------
__device__ __forceinline__ void fill_b128(_Float16* Blds, const float* __restrict__ W,
                                          int ncols_valid, int ldw, int kmax,
                                          int tid, int nth) {
  for (int idx = tid; idx < 16384; idx += nth) {
    int j    = idx & 15;
    int col  = (idx >> 4) & 127;
    int half = (idx >> 11) & 1;
    int kb   = idx >> 12;
    int k    = kb * 32 + half * 8 + (j & 7) + (j >> 3) * 16;
    float v  = (col < ncols_valid && k < kmax) ? W[(long)col * ldw + k] : 0.0f;
    Blds[idx] = (_Float16)v;
  }
}

__device__ __forceinline__ v16h load_b_frag(const _Float16* Blds, int kb, int half,
                                            int nt, int col) {
  return *(const v16h*)(Blds + ((((kb * 2 + half) * 128) + nt * 16 + col) << 4));
}

// ---------------------------------------------------------------------------
// Big edge GEMM: nbr1[E,128] (f16) = nbr[E,F] @ W1x[128,F]^T (WMMA f16->f32).
// 8 waves/block, 16 rows/wave -> 128 rows/block.
// ---------------------------------------------------------------------------
template <int KMAX>
__global__ __launch_bounds__(256) void k_gemm_nbr1(const float* __restrict__ nbr,
                                                   const float* __restrict__ W1x,
                                                   _Float16* __restrict__ nbr1,
                                                   long E, int F) {
  __shared__ __align__(32) _Float16 Blds[16384];
  fill_b128(Blds, W1x, 128, F, F, threadIdx.x, 256);
  __syncthreads();

  int lane = threadIdx.x & 31, wave = threadIdx.x >> 5;
  int half = lane >> 4, col = lane & 15;
  long rowBase = (long)blockIdx.x * 128 + wave * 16;
  long arow = rowBase + col;
  long rclamp = arow < E ? arow : (E - 1);      // clamp: no load guards needed
  const float* p = nbr + rclamp * (long)F;

  __builtin_prefetch(p + 128 * (long)F, 0, 1);

  v8f acc[8] = {};
#pragma unroll
  for (int kb = 0; kb < 4; ++kb) {
    v16h a = load_a_f32<KMAX>(p, kb, half);
#pragma unroll
    for (int nt = 0; nt < 8; ++nt) {
      v16h b = load_b_frag(Blds, kb, half, nt, col);
      acc[nt] = __builtin_amdgcn_wmma_f32_16x16x32_f16(false, a, false, b,
                                                       (short)0, acc[nt], false, false);
    }
  }
  long orow = rowBase + half * 8;
#pragma unroll
  for (int nt = 0; nt < 8; ++nt)
#pragma unroll
    for (int g = 0; g < 8; ++g) {
      long r = orow + g;
      if (r < E) nbr1[r * 128 + nt * 16 + col] = (_Float16)acc[nt][g];
    }
}

// ---------------------------------------------------------------------------
// Dual GEMM: out[M,128] (f32) = A0@W0^T + A1@W1^T,  A* [M,KMAX] (ld=KMAX),
// W* [128,KMAX]. Layer1: (x,f | W1x,W1n); Layer2: (h1,f2 | W2x,W2n).
// ---------------------------------------------------------------------------
template <int KMAX>
__global__ __launch_bounds__(256) void k_gemm_dual(const float* __restrict__ A0,
                                                   const float* __restrict__ A1,
                                                   const float* __restrict__ W0,
                                                   const float* __restrict__ W1,
                                                   float* __restrict__ out, long M) {
  __shared__ __align__(32) _Float16 B0[16384];
  __shared__ __align__(32) _Float16 B1[16384];
  fill_b128(B0, W0, 128, KMAX, KMAX, threadIdx.x, 256);
  fill_b128(B1, W1, 128, KMAX, KMAX, threadIdx.x, 256);
  __syncthreads();

  int lane = threadIdx.x & 31, wave = threadIdx.x >> 5;
  int half = lane >> 4, col = lane & 15;
  long rowBase = (long)blockIdx.x * 128 + wave * 16;
  long arow = rowBase + col;
  long rclamp = arow < M ? arow : (M - 1);

  const float*    As[2] = {A0 + rclamp * (long)KMAX, A1 + rclamp * (long)KMAX};
  const _Float16* Bs[2] = {B0, B1};

  v8f acc[8] = {};
#pragma unroll
  for (int s = 0; s < 2; ++s) {
    __builtin_prefetch(As[s] + 128 * (long)KMAX, 0, 1);
#pragma unroll
    for (int kb = 0; kb < 4; ++kb) {
      v16h a = load_a_f32<KMAX>(As[s], kb, half);
#pragma unroll
      for (int nt = 0; nt < 8; ++nt) {
        v16h b = load_b_frag(Bs[s], kb, half, nt, col);
        acc[nt] = __builtin_amdgcn_wmma_f32_16x16x32_f16(false, a, false, b,
                                                         (short)0, acc[nt], false, false);
      }
    }
  }
  long orow = rowBase + half * 8;
#pragma unroll
  for (int nt = 0; nt < 8; ++nt)
#pragma unroll
    for (int g = 0; g < 8; ++g) {
      long r = orow + g;
      if (r < M) out[r * 128 + nt * 16 + col] = acc[nt][g];
    }
}

// ---------------------------------------------------------------------------
// Classifier: out[M,C] = relu(BN(h2)) @ Wc[C,128]^T + bc.  BN applied in
// registers while building A-fragments; Wc padded to 48 cols (3 N-tiles).
// ---------------------------------------------------------------------------
__global__ __launch_bounds__(256) void k_classifier(const float* __restrict__ h2,
                                                    const float* __restrict__ Wc,
                                                    const float* __restrict__ bc,
                                                    float* __restrict__ out,
                                                    long M, const float* __restrict__ stats,
                                                    float cntElems,
                                                    const float* __restrict__ gp,
                                                    const float* __restrict__ bp, int C) {
  __shared__ __align__(32) _Float16 Blds[6144]; // 4 kb * 2 half * 48 col * 16 j
  for (int idx = threadIdx.x; idx < 6144; idx += 256) {
    int j   = idx & 15;
    int col = (idx >> 4) % 48;
    int rem = idx / 768;            // kb*2 + half
    int half = rem & 1, kb = rem >> 1;
    int k = kb * 32 + half * 8 + (j & 7) + (j >> 3) * 16;
    float v = (col < C) ? Wc[(long)col * 128 + k] : 0.0f;
    Blds[idx] = (_Float16)v;
  }
  __syncthreads();

  float m   = stats[0] / cntElems;
  float var = stats[1] / cntElems - m * m;
  float inv = rsqrtf(var + BN_EPS);
  float g = gp[0], b = bp[0];

  int lane = threadIdx.x & 31, wave = threadIdx.x >> 5;
  int half = lane >> 4, col = lane & 15;
  long rowBase = (long)blockIdx.x * 128 + wave * 16;
  long arow = rowBase + col;
  long rclamp = arow < M ? arow : (M - 1);
  const float* p = h2 + rclamp * 128;

  v8f acc[3] = {};
#pragma unroll
  for (int kb = 0; kb < 4; ++kb) {
    _Float16 t[16];
    int k0 = kb * 32 + half * 8;
#pragma unroll
    for (int c = 0; c < 4; ++c) {                   // 4 float4 chunks
      int cs = k0 + (c & 1) * 4 + (c >> 1) * 16;
      float4 v4 = *(const float4*)(p + cs);
      float vv[4] = {v4.x, v4.y, v4.z, v4.w};
#pragma unroll
      for (int i = 0; i < 4; ++i)
        t[c * 4 + i] = (_Float16)fmaxf(0.0f, g * (vv[i] - m) * inv + b);
    }
    v16h a;
#pragma unroll
    for (int i = 0; i < 16; ++i) a[i] = t[i];
#pragma unroll
    for (int nt = 0; nt < 3; ++nt) {
      v16h bf = *(const v16h*)(Blds + ((((kb * 2 + half) * 48) + nt * 16 + col) << 4));
      acc[nt] = __builtin_amdgcn_wmma_f32_16x16x32_f16(false, a, false, bf,
                                                       (short)0, acc[nt], false, false);
    }
  }
  long orow = rowBase + half * 8;
#pragma unroll
  for (int nt = 0; nt < 3; ++nt)
#pragma unroll
    for (int gg = 0; gg < 8; ++gg) {
      long r = orow + gg;
      int cg = nt * 16 + col;
      if (r < M && cg < C) out[r * C + cg] = acc[nt][gg] + bc[cg];
    }
}

// ---------------------------------------------------------------------------
// seg_ids is sorted: off[n] = lower_bound(seg_ids, n), n in [0, N].
// ---------------------------------------------------------------------------
__global__ void k_offsets(const int* __restrict__ ids, int* __restrict__ off,
                          int N, int E) {
  int n = blockIdx.x * blockDim.x + threadIdx.x;
  if (n > N) return;
  int lo = 0, hi = E;
  while (lo < hi) {
    int mid = (lo + hi) >> 1;
    if (ids[mid] < n) lo = mid + 1; else hi = mid;
  }
  off[n] = lo;
}

// Per-node neighbor mean of raw features: f[n,0:F].
__global__ __launch_bounds__(128) void k_seg_mean_f(const float* __restrict__ nbr,
                                                    const int* __restrict__ off,
                                                    float* __restrict__ f, int F) {
  int n = blockIdx.x, t = threadIdx.x;
  int e0 = off[n], e1 = off[n + 1];
  if (t >= F) return;
  float s = 0.0f;
  for (int e = e0; e < e1; ++e) s += nbr[(long)e * F + t];
  f[(long)n * F + t] = s / fmaxf((float)(e1 - e0), 1.0f);
}

// Fused per-segment BN(stats over k_i*128 elems) + ReLU + mean -> f2[n,0:128].
// Two passes over the (mostly L2-resident) f16 nbr1 rows of this node.
__global__ __launch_bounds__(128) void k_seg_bn_f2(const _Float16* __restrict__ nbr1,
                                                   const int* __restrict__ off,
                                                   const float* __restrict__ gp,
                                                   const float* __restrict__ bp,
                                                   float* __restrict__ f2) {
  __shared__ float ss[128], sq[128], bcast[2];
  int n = blockIdx.x, t = threadIdx.x;
  int e0 = off[n], e1 = off[n + 1];
  int cnt = e1 - e0;

  float s = 0.0f, q = 0.0f;
  for (int e = e0; e < e1; ++e) {
    float v = (float)nbr1[(long)e * 128 + t];
    s += v; q += v * v;
  }
  ss[t] = s; sq[t] = q;
  __syncthreads();
  for (int o = 64; o > 0; o >>= 1) {
    if (t < o) { ss[t] += ss[t + o]; sq[t] += sq[t + o]; }
    __syncthreads();
  }
  if (t == 0) {
    float ne  = fmaxf((float)cnt * 128.0f, 1.0f);
    float sm  = ss[0] / ne;
    float var = sq[0] / ne - sm * sm;
    bcast[0] = sm;
    bcast[1] = rsqrtf(var + BN_EPS);
  }
  __syncthreads();
  float sm = bcast[0], inv = bcast[1], g = gp[0], b = bp[0];
  float acc = 0.0f;
  for (int e = e0; e < e1; ++e) {
    float v = (float)nbr1[(long)e * 128 + t];
    acc += fmaxf(0.0f, g * (v - sm) * inv + b);
  }
  f2[(long)n * 128 + t] = acc / fmaxf((float)cnt, 1.0f);
}

// Global sum / sum-of-squares reduction (for scalar BatchNorm stats).
__global__ __launch_bounds__(256) void k_reduce(const float* __restrict__ src, long n,
                                                float* __restrict__ dst) {
  __shared__ float ss[256], sq[256];
  float s = 0.0f, q = 0.0f;
  for (long i = blockIdx.x * (long)blockDim.x + threadIdx.x; i < n;
       i += (long)gridDim.x * blockDim.x) {
    float v = src[i];
    s += v; q += v * v;
  }
  ss[threadIdx.x] = s; sq[threadIdx.x] = q;
  __syncthreads();
  for (int o = 128; o > 0; o >>= 1) {
    if (threadIdx.x < o) { ss[threadIdx.x] += ss[threadIdx.x + o]; sq[threadIdx.x] += sq[threadIdx.x + o]; }
    __syncthreads();
  }
  if (threadIdx.x == 0) { atomicAdd(dst, ss[0]); atomicAdd(dst + 1, sq[0]); }
}

// In-place scalar BN + ReLU.
__global__ __launch_bounds__(256) void k_bn_relu(float* __restrict__ h, long n,
                                                 const float* __restrict__ stats,
                                                 float cnt,
                                                 const float* __restrict__ gp,
                                                 const float* __restrict__ bp) {
  float m   = stats[0] / cnt;
  float var = stats[1] / cnt - m * m;
  float inv = rsqrtf(var + BN_EPS);
  float g = gp[0], b = bp[0];
  for (long i = blockIdx.x * (long)blockDim.x + threadIdx.x; i < n;
       i += (long)gridDim.x * blockDim.x)
    h[i] = fmaxf(0.0f, g * (h[i] - m) * inv + b);
}

__global__ void k_zero4(float* p) { if (threadIdx.x < 4) p[threadIdx.x] = 0.0f; }

// ---------------------------------------------------------------------------
extern "C" void kernel_launch(void* const* d_in, const int* in_sizes, int n_in,
                              void* d_out, int out_size, void* d_ws, size_t ws_size,
                              hipStream_t stream) {
  const float* x   = (const float*)d_in[0];
  const float* nbr = (const float*)d_in[1];
  const int*   seg = (const int*)  d_in[2];
  const float* W1x = (const float*)d_in[3];
  const float* W1n = (const float*)d_in[4];
  const float* g1  = (const float*)d_in[5];
  const float* b1  = (const float*)d_in[6];
  const float* W2x = (const float*)d_in[7];
  const float* W2n = (const float*)d_in[8];
  const float* g2  = (const float*)d_in[9];
  const float* b2  = (const float*)d_in[10];
  const float* Wc  = (const float*)d_in[11];
  const float* bc  = (const float*)d_in[12];

  long E = in_sizes[2];
  int  F = (int)(in_sizes[1] / E);    // 100
  long N = in_sizes[0] / F;
  int  C = in_sizes[12];              // 40

  // Workspace carve-up (256B aligned).
  char* cur = (char*)d_ws;
  auto take = [&](size_t bytes) -> void* {
    void* p = (void*)cur;
    cur += (bytes + 255) & ~(size_t)255;
    return p;
  };
  int*      off  = (int*)     take((size_t)(N + 1) * sizeof(int));
  float*    scal = (float*)   take(4 * sizeof(float));            // [s1,q1,s2,q2]
  float*    f    = (float*)   take((size_t)N * F * sizeof(float));
  float*    h1   = (float*)   take((size_t)N * 128 * sizeof(float));
  float*    f2   = (float*)   take((size_t)N * 128 * sizeof(float));
  float*    h2   = (float*)   take((size_t)N * 128 * sizeof(float));
  _Float16* nbr1 = (_Float16*)take((size_t)E * 128 * sizeof(_Float16));

  long nH = N * 128;

  k_zero4<<<1, 32, 0, stream>>>(scal);
  k_offsets<<<(int)((N + 1 + 255) / 256), 256, 0, stream>>>(seg, off, (int)N, (int)E);
  k_seg_mean_f<<<(int)N, 128, 0, stream>>>(nbr, off, f, F);
  k_gemm_nbr1<100><<<(int)((E + 127) / 128), 256, 0, stream>>>(nbr, W1x, nbr1, E, F);
  k_gemm_dual<100><<<(int)((N + 127) / 128), 256, 0, stream>>>(x, f, W1x, W1n, h1, N);
  k_reduce<<<512, 256, 0, stream>>>(h1, nH, scal);
  k_bn_relu<<<1024, 256, 0, stream>>>(h1, nH, scal, (float)nH, g1, b1);
  k_seg_bn_f2<<<(int)N, 128, 0, stream>>>(nbr1, off, g1, b1, f2);
  k_gemm_dual<128><<<(int)((N + 127) / 128), 256, 0, stream>>>(h1, f2, W2x, W2n, h2, N);
  k_reduce<<<512, 256, 0, stream>>>(h2, nH, scal + 2);
  k_classifier<<<(int)((N + 127) / 128), 256, 0, stream>>>(h2, Wc, bc, (float*)d_out,
                                                           N, scal + 2, (float)nH,
                                                           g2, b2, C);
}
